// RhnModel_67559835566850
// MI455X (gfx1250) — matrix operand order repaired and checked by reading, not compiled, not run on hardware
//
#include <hip/hip_runtime.h>

#define T_ 512
#define B_ 64
#define V_ 256
#define E_ 512
#define H_ 512
#define L_ 2
#define D_ 4

typedef __attribute__((ext_vector_type(16))) __bf16 v16bf;
typedef __attribute__((ext_vector_type(8)))  __bf16 v8bf;
typedef __attribute__((ext_vector_type(8)))  float  v8f;

// Build a 16x32 bf16 A-fragment for one lane from a row-major [16][512] bf16 buffer.
// Lane L (row = L&15) holds K = 2*kb..2*kb+15 (contiguous, matches packed B), kb = (L>>4)*8.
__device__ __forceinline__ v16bf load_a_frag(const __bf16* rowptr, int kb) {
  v8bf lo = *(const v8bf*)(rowptr + 2 * kb);
  v8bf hi = *(const v8bf*)(rowptr + 2 * kb + 8);
  return __builtin_shufflevector(lo, hi, 0,1,2,3,4,5,6,7,8,9,10,11,12,13,14,15);
}

__device__ __forceinline__ v8f splat8(float v) {
  v8f r = {v, v, v, v, v, v, v, v};
  return r;
}

// ---------------------------------------------------------------------------
// Pack fp32 row-major [K][N] weight matrices into bf16 32x16 B-fragments,
// fragment order [ntile][ktile], element order = per-lane layout used here (contiguous K, as load_a_frag):
//   lane = ((k&31)>=16 ? 16 : 0) + (n&15),  16 consecutive K values per lane.
// ---------------------------------------------------------------------------
__global__ void pack_b_frags(const float* __restrict__ src, __bf16* __restrict__ dst,
                             int K, int N, int nMat) {
  const int matElems = K * N;
  const int nK = K >> 5;
  const size_t total = (size_t)matElems * nMat;
  for (size_t e = (size_t)blockIdx.x * blockDim.x + threadIdx.x; e < total;
       e += (size_t)gridDim.x * blockDim.x) {
    int m       = (int)(e / (size_t)matElems);
    int em      = (int)(e % (size_t)matElems);
    int fragIdx = em >> 9;
    int within  = em & 511;
    int lane    = within >> 4;
    int i       = within & 15;
    int kt      = fragIdx % nK;
    int ntile   = fragIdx / nK;
    int k       = kt * 32 + (lane >> 4) * 16 + i;
    int n       = ntile * 16 + (lane & 15);
    dst[e] = (__bf16)src[(size_t)m * matElems + (size_t)k * N + n];
  }
}

// ---------------------------------------------------------------------------
// Persistent recurrence kernel: one workgroup (16 waves) per 16-row batch slice.
// fp32 state master lives in registers (each lane permanently owns 2x8 elems
// per layer); bf16 shadow in LDS feeds WMMA A-fragments. Wave w owns h-cols
// [32w,32w+32) and g-cols [512+32w, 512+32w+32) so the highway combine is
// lane-local. Per-timestep output leaves via async LDS->global DMA (ASYNCcnt),
// overlapped with the next timestep's embedding gather + layer-0 compute.
// ---------------------------------------------------------------------------
__launch_bounds__(512, 1)
__global__ void rhn_recurrence(const int*   __restrict__ x,
                               const float* __restrict__ emb,
                               const __bf16* __restrict__ whF,
                               const __bf16* __restrict__ winF,
                               const float* __restrict__ b_h,
                               __bf16*      __restrict__ outB,
                               float*       __restrict__ sfin) {
  __shared__ __align__(16) __bf16 s_bf[2][16 * 512];   // per-layer bf16 state
  __shared__ __align__(16) __bf16 inp_bf[16 * 512];    // timestep input (embedding)
  __shared__ int tok[16];

  const int tid   = threadIdx.x;
  const int wave  = tid >> 5;
  const int lane  = tid & 31;
  const int slice = blockIdx.x;        // batch rows [16*slice, 16*slice+16)
  const int lhalf = lane >> 4;
  const int l15   = lane & 15;
  const int arow  = l15;               // A-fragment row for this lane
  const int kb    = lhalf * 8;         // A-fragment K sub-base

  for (int e = tid; e < 16 * 512; e += 512) {
    s_bf[0][e] = (__bf16)0.f;
    s_bf[1][e] = (__bf16)0.f;
  }
  __syncthreads();

  float sreg[2][2][8];
#pragma unroll
  for (int l = 0; l < 2; ++l)
#pragma unroll
    for (int j = 0; j < 2; ++j)
#pragma unroll
      for (int r = 0; r < 8; ++r) sreg[l][j][r] = 0.f;

  for (int t = 0; t < T_; ++t) {
    // -------- gather embedding rows for this slice into LDS (bf16) --------
    if (tid < 16) tok[tid] = x[t * B_ + slice * 16 + tid];
    __syncthreads();
    for (int e = tid; e < 16 * 512; e += 512) {
      int row = e >> 9, col = e & 511;
      inp_bf[e] = (__bf16)emb[(size_t)tok[row] * E_ + col];
    }
    __syncthreads();

#pragma unroll
    for (int l = 0; l < L_; ++l) {
      const __bf16* inp = (l == 0) ? inp_bf : s_bf[0];
      __bf16* sb = s_bf[l];
      if (l == 1) {
        // Pending async stores of last timestep read s_bf[1]; they must drain
        // before this layer's elementwise writes. This wait precedes (in
        // program order) the post-GEMM barrier that gates the first write,
        // so every thread's DMA is complete before any thread overwrites.
        asm volatile("s_wait_asynccnt 0x0" ::: "memory");
      }
      for (int d = 0; d < D_; ++d) {
        // accumulators: j=0,1 -> h tiles; j=2,3 -> g tiles (same cols + 512)
        v8f acc[4];
#pragma unroll
        for (int j = 0; j < 4; ++j) {
          int ntile = (j < 2) ? (wave * 2 + j) : (32 + wave * 2 + (j - 2));
          acc[j] = splat8(b_h[(size_t)(l * D_ + d) * 1024 + ntile * 16 + l15]);
        }
        const __bf16* wmat = whF + (size_t)(l * D_ + d) * (512 * 1024);
        for (int kt = 0; kt < 16; ++kt) {
          v16bf a = load_a_frag(&sb[arow * 512 + kt * 32], kb);
          __builtin_prefetch(wmat + (((size_t)(wave * 2) * 16 + kt + 1) << 9), 0, 1);
#pragma unroll
          for (int j = 0; j < 4; ++j) {
            int ntile = (j < 2) ? (wave * 2 + j) : (32 + wave * 2 + (j - 2));
            v16bf b = *(const v16bf*)(wmat + ((size_t)(ntile * 16 + kt) << 9) + lane * 16);
            acc[j] = __builtin_amdgcn_wmma_f32_16x16x32_bf16(
                false, a, false, b, (short)0, acc[j], false, false);
          }
        }
        if (d == 0) {  // fold inp @ w_in into the first micro-step
          const __bf16* wmat2 = winF + (size_t)l * (512 * 1024);
          for (int kt = 0; kt < 16; ++kt) {
            v16bf a = load_a_frag(&inp[arow * 512 + kt * 32], kb);
#pragma unroll
            for (int j = 0; j < 4; ++j) {
              int ntile = (j < 2) ? (wave * 2 + j) : (32 + wave * 2 + (j - 2));
              v16bf b = *(const v16bf*)(wmat2 + ((size_t)(ntile * 16 + kt) << 9) + lane * 16);
              acc[j] = __builtin_amdgcn_wmma_f32_16x16x32_bf16(
                  false, a, false, b, (short)0, acc[j], false, false);
            }
          }
        }
        __syncthreads();  // all waves done reading sb this micro-step
        // highway update: s = tanh(h)*sig(g) + s*(1-sig(g)); lane-local
#pragma unroll
        for (int j = 0; j < 2; ++j) {
          int col = wave * 32 + j * 16 + l15;
#pragma unroll
          for (int r = 0; r < 8; ++r) {
            float h  = tanhf(acc[j][r]);
            float g  = 1.f / (1.f + __expf(-acc[j + 2][r]));
            float sn = h * g + sreg[l][j][r] * (1.f - g);
            sreg[l][j][r] = sn;
            sb[(lhalf * 8 + r) * 512 + col] = (__bf16)sn;  // C-layout row = r + 8*lhalf
          }
        }
        __syncthreads();  // new state visible to all waves
      }
    }
    // Stream layer-1 state (this timestep's output) to global via async
    // LDS->global DMA: no VGPR round-trip, completion deferred to next
    // timestep's layer-1 (overlaps gather + layer-0 GEMMs).
    {
      const size_t gbase = ((size_t)t * B_ + slice * 16) * H_;
#pragma unroll
      for (int c = 0; c < 2; ++c) {
        int e = (tid + c * 512) * 8;                       // 16B chunk per store
        unsigned lds_addr = (unsigned)(uintptr_t)(&s_bf[1][e]);
        unsigned long long gaddr = (unsigned long long)(uintptr_t)(outB + gbase + e);
        asm volatile("global_store_async_from_lds_b128 %0, %1, off"
                     :: "v"(gaddr), "v"(lds_addr) : "memory");
      }
    }
  }
  asm volatile("s_wait_asynccnt 0x0" ::: "memory");

  // final states (fp32, from registers) -> d_out tail [L,B,H]
#pragma unroll
  for (int l = 0; l < 2; ++l)
#pragma unroll
    for (int j = 0; j < 2; ++j) {
      int col = wave * 32 + j * 16 + l15;
#pragma unroll
      for (int r = 0; r < 8; ++r) {
        int row = slice * 16 + lhalf * 8 + r;
        sfin[(size_t)l * B_ * H_ + (size_t)row * H_ + col] = sreg[l][j][r];
      }
    }
}

// ---------------------------------------------------------------------------
// Logits GEMM: [T*B=32768, 512] @ [512, 256] + bias. One wave per 16x64 tile.
// ---------------------------------------------------------------------------
__launch_bounds__(256, 2)
__global__ void rhn_logits(const __bf16* __restrict__ outB,
                           const __bf16* __restrict__ wfcF,
                           const float*  __restrict__ b_fc,
                           float*        __restrict__ logits) {
  const int tid   = threadIdx.x;
  const int lane  = tid & 31;
  const int l15   = lane & 15;
  const int lhalf = lane >> 4;
  const int gw    = blockIdx.x * 8 + (tid >> 5);
  const int mtile = gw >> 2;   // 0..2047
  const int ng    = gw & 3;    // 64-col group

  v8f acc[4];
#pragma unroll
  for (int j = 0; j < 4; ++j) acc[j] = splat8(b_fc[(ng * 4 + j) * 16 + l15]);

  const __bf16* arow = outB + (size_t)(mtile * 16 + l15) * H_;
  const int kb = lhalf * 8;
  for (int kt = 0; kt < 16; ++kt) {
    v16bf a = load_a_frag(arow + kt * 32, kb);
#pragma unroll
    for (int j = 0; j < 4; ++j) {
      int ntile = ng * 4 + j;
      v16bf b = *(const v16bf*)(wfcF + ((size_t)(ntile * 16 + kt) << 9) + lane * 16);
      acc[j] = __builtin_amdgcn_wmma_f32_16x16x32_bf16(
          false, a, false, b, (short)0, acc[j], false, false);
    }
  }
#pragma unroll
  for (int j = 0; j < 4; ++j) {
    int col = (ng * 4 + j) * 16 + l15;
#pragma unroll
    for (int r = 0; r < 8; ++r) {
      int row = mtile * 16 + lhalf * 8 + r;
      logits[(size_t)row * V_ + col] = acc[j][r];
    }
  }
}

extern "C" void kernel_launch(void* const* d_in, const int* in_sizes, int n_in,
                              void* d_out, int out_size, void* d_ws, size_t ws_size,
                              hipStream_t stream) {
  const int*   x    = (const int*)d_in[0];
  const float* emb  = (const float*)d_in[1];
  const float* w_in = (const float*)d_in[2];
  const float* w_h  = (const float*)d_in[3];
  const float* b_h  = (const float*)d_in[4];
  const float* w_fc = (const float*)d_in[5];
  const float* b_fc = (const float*)d_in[6];

  float* logits = (float*)d_out;                       // [T,B,V]
  float* sfin   = logits + (size_t)T_ * B_ * V_;       // [L,B,H]

  // workspace layout (bf16): packed weights + timestep output buffer (~44 MB)
  __bf16* whF  = (__bf16*)d_ws;                        // L*D*512*1024
  __bf16* winF = whF + (size_t)L_ * D_ * 512 * 1024;   // L*512*1024
  __bf16* wfcF = winF + (size_t)L_ * 512 * 1024;       // 512*256
  __bf16* outB = wfcF + (size_t)512 * 256;             // T*B*H

  pack_b_frags<<<512, 256, 0, stream>>>(w_h,  whF,  512, 1024, L_ * D_);
  pack_b_frags<<<256, 256, 0, stream>>>(w_in, winF, 512, 1024, L_);
  pack_b_frags<<<64,  256, 0, stream>>>(w_fc, wfcF, 512, 256,  1);
  rhn_recurrence<<<4, 512, 0, stream>>>(x, emb, whF, winF, b_h, outB, sfin);
  rhn_logits<<<1024, 256, 0, stream>>>(outB, wfcF, b_fc, logits);
}
